// ResidualAttentionBlock_6502580486477
// MI455X (gfx1250) — matrix operand order repaired and checked by reading, not compiled, not run on hardware
//
#include <hip/hip_runtime.h>
#include <hip/hip_bf16.h>

// ---------------------------------------------------------------------------
// MI455X (gfx1250) residual attention block, bf16 WMMA pipeline.
// D = 1024, H = 16, ch = 64, B = 8, S = 1024, rows M = 8192.
// ---------------------------------------------------------------------------

typedef __attribute__((ext_vector_type(16))) __bf16 v16bf;
typedef __attribute__((ext_vector_type(8)))  float  v8f;
typedef unsigned int v4u __attribute__((ext_vector_type(4)));
typedef int          v8i __attribute__((ext_vector_type(8)));
typedef int          v4i __attribute__((ext_vector_type(4)));

#define HAVE_TDM __has_builtin(__builtin_amdgcn_tensor_load_to_lds)

union BF16Frag { unsigned int u[8]; v16bf v; };

__device__ __forceinline__ unsigned short f2bf(float f) {
  unsigned int u = __float_as_uint(f);
  u = (u + 0x7FFFu + ((u >> 16) & 1u)) >> 16;   // round-to-nearest-even
  return (unsigned short)u;
}

// ---------------------------------------------------------------------------
// Tiled GEMM: C[M,N] = act(A[M,K] * B[K,N] + bias) (+ residual)
//   AS = float  : A is fp32, converted to bf16 while staging to LDS
//   AS = ushort : A is already bf16
//   MODE 0: store bf16 | MODE 1: QuickGELU -> bf16 | MODE 2: +residual -> fp32
// Block = 256 threads (8 waves). Macro tile 128x128, K-step 32.
// Wave (wm,wn) in 4x2 grid owns a 32(row) x 64(col) sub-tile = 8 WMMA accs.
// ---------------------------------------------------------------------------
template <typename AS, int MODE>
__global__ __launch_bounds__(256) void gemm_bf16_wmma(
    const AS* A, const float* Bw, const float* bias,
    void* Cout, const float* Res, int M, int N, int K)
{
  __shared__ unsigned short As[128 * 40];   // [row][k], stride 40 (20 banks)
  __shared__ unsigned short Bs[128 * 40];   // [col][k], stride 40

  const int tid  = threadIdx.x;
  const int lane = tid & 31;
  const int wid  = tid >> 5;
  const int hf   = lane >> 4;    // lane half (C rows 0-7 vs 8-15)
  const int l16  = lane & 15;
  const int wm   = wid & 3;
  const int wn   = wid >> 2;
  const long m0 = (long)blockIdx.y * 128;
  const long n0 = (long)blockIdx.x * 128;

  v8f acc[2][4];
  #pragma unroll
  for (int i = 0; i < 2; ++i)
    #pragma unroll
    for (int j = 0; j < 4; ++j)
      #pragma unroll
      for (int r = 0; r < 8; ++r) acc[i][j][r] = 0.0f;

  for (int kb = 0; kb < K; kb += 32) {
    __syncthreads();
    // ---- stage A tile 128x32 (vectorized 16B/8B loads, bf16 into LDS) ----
    #pragma unroll
    for (int i = 0; i < 4; ++i) {
      const int idx = tid + i * 256;
      const int r   = idx >> 3;
      const int c4  = (idx & 7) * 4;
      unsigned int p0, p1;
      if constexpr (sizeof(AS) == 4) {
        const float4 f = *reinterpret_cast<const float4*>(
            (const float*)A + (size_t)(m0 + r) * K + kb + c4);
        p0 = (unsigned int)f2bf(f.x) | ((unsigned int)f2bf(f.y) << 16);
        p1 = (unsigned int)f2bf(f.z) | ((unsigned int)f2bf(f.w) << 16);
      } else {
        const uint2 d = *reinterpret_cast<const uint2*>(
            (const unsigned short*)A + (size_t)(m0 + r) * K + kb + c4);
        p0 = d.x; p1 = d.y;
      }
      *reinterpret_cast<uint2*>(&As[r * 40 + c4]) = make_uint2(p0, p1);
    }
    // ---- stage B tile 32x128 transposed -> Bs[n][k] ----
    #pragma unroll
    for (int i = 0; i < 4; ++i) {
      const int idx = tid + i * 256;
      const int kr  = idx >> 5;
      const int n4  = (idx & 31) * 4;
      const float4 f = *reinterpret_cast<const float4*>(
          Bw + (size_t)(kb + kr) * N + n0 + n4);
      Bs[(n4 + 0) * 40 + kr] = f2bf(f.x);
      Bs[(n4 + 1) * 40 + kr] = f2bf(f.y);
      Bs[(n4 + 2) * 40 + kr] = f2bf(f.z);
      Bs[(n4 + 3) * 40 + kr] = f2bf(f.w);
    }
    // ---- prefetch next K-step tiles (global_prefetch, L2 -> WGP) ----
    if (kb + 32 < K) {
      const int ra = tid >> 3, ca = (tid & 7) * 4;
      __builtin_prefetch((const char*)A + ((size_t)(m0 + ra) * K + kb + 32 + ca) * sizeof(AS), 0, 1);
      const int rb = tid >> 5, nb = (tid & 31) * 4;
      __builtin_prefetch(Bw + (size_t)(kb + 32 + rb) * N + n0 + nb, 0, 1);
    }
    __syncthreads();

    // ---- gather WMMA fragments (ISA 16-bit A/B VGPR layouts) ----
    BF16Frag a[2], b[4];
    #pragma unroll
    for (int i = 0; i < 2; ++i) {
      const int row = (wm * 2 + i) * 16 + l16;
      #pragma unroll
      for (int v = 0; v < 8; ++v) {
        const int kk = ((v & 4) ? 16 : 0) + hf * 8 + 2 * (v & 3);
        a[i].u[v] = *reinterpret_cast<const unsigned int*>(&As[row * 40 + kk]);
      }
    }
    #pragma unroll
    for (int j = 0; j < 4; ++j) {
      const int col = (wn * 4 + j) * 16 + l16;
      #pragma unroll
      for (int v = 0; v < 8; ++v) {
        const int kk = ((v & 4) ? 16 : 0) + hf * 8 + 2 * (v & 3);
        b[j].u[v] = *reinterpret_cast<const unsigned int*>(&Bs[col * 40 + kk]);
      }
    }
    #pragma unroll
    for (int i = 0; i < 2; ++i)
      #pragma unroll
      for (int j = 0; j < 4; ++j)
        acc[i][j] = __builtin_amdgcn_wmma_f32_16x16x32_bf16(
            false, a[i].v, false, b[j].v, (short)0, acc[i][j], false, false);
  }

  // ---- epilogue: bias (+activation / +residual) ----
  #pragma unroll
  for (int i = 0; i < 2; ++i) {
    #pragma unroll
    for (int j = 0; j < 4; ++j) {
      const long gn = n0 + (wn * 4 + j) * 16 + l16;
      const float bv = bias[gn];
      #pragma unroll
      for (int r = 0; r < 8; ++r) {
        const long gm = m0 + (wm * 2 + i) * 16 + hf * 8 + r;
        float val = acc[i][j][r] + bv;
        if constexpr (MODE == 0) {
          ((unsigned short*)Cout)[gm * N + gn] = f2bf(val);
        } else if constexpr (MODE == 1) {
          val = val / (1.0f + __expf(-1.702f * val));   // QuickGELU
          ((unsigned short*)Cout)[gm * N + gn] = f2bf(val);
        } else {
          ((float*)Cout)[gm * N + gn] = val + Res[gm * N + gn];
        }
      }
    }
  }
}

// ---------------------------------------------------------------------------
// Flash-style attention with diagonal -10 mask, fused x + attn residual.
// Grid: (q_tile 0..7, head 0..15, batch 0..7), 256 threads (8 waves).
// Wave w owns query rows [q0 + 16w, q0 + 16w + 16). Key blocks of 64.
// QKV layout (from QKV GEMM): bf16 [b*S + s][h*192 + {q:0, k:64, v:128} + c].
// K tile staged by the Tensor Data Mover (tensor_load_to_lds): 64x64 bf16
// tile, 128B rows padded to 136B in LDS (pad_interval=32 DW, pad_amount=2 DW).
// ---------------------------------------------------------------------------
__global__ __launch_bounds__(256) void attn_flash_wmma(
    const unsigned short* __restrict__ QKV,
    const float* __restrict__ X,
    float* __restrict__ Out)
{
  __shared__ unsigned short Ks[64 * 68];      // [s_key][c]  (B for Q·K^T)
  __shared__ unsigned short Vs[64 * 68];      // [c][s_key]  (B for P·V)
  __shared__ unsigned short Pl[8][16 * 68];   // per-wave P (A for P·V)

  const int tid  = threadIdx.x;
  const int lane = tid & 31;
  const int wid  = tid >> 5;
  const int hf   = lane >> 4;
  const int l16  = lane & 15;
  const int q0   = blockIdx.x * 128;
  const int h    = blockIdx.y;
  const int b    = blockIdx.z;
  const size_t baseRow = (size_t)b * 1024;

  // ---- Q fragments: 16 query rows x 64 ch, 2 K-steps, kept in VGPRs ----
  BF16Frag qf[2];
  {
    const int qrow = q0 + wid * 16 + l16;
    const unsigned short* qp = QKV + (baseRow + qrow) * 3072 + h * 192;
    #pragma unroll
    for (int ks = 0; ks < 2; ++ks)
      #pragma unroll
      for (int v = 0; v < 8; ++v) {
        const int kk = ks * 32 + ((v & 4) ? 16 : 0) + hf * 8 + 2 * (v & 3);
        qf[ks].u[v] = *reinterpret_cast<const unsigned int*>(qp + kk);
      }
  }

  float mrow[8], lrow[8];
  v8f Oa[4];
  #pragma unroll
  for (int r = 0; r < 8; ++r) { mrow[r] = -1e30f; lrow[r] = 0.0f; }
  #pragma unroll
  for (int j = 0; j < 4; ++j)
    #pragma unroll
    for (int r = 0; r < 8; ++r) Oa[j][r] = 0.0f;

  for (int kb0 = 0; kb0 < 1024; kb0 += 64) {
    __syncthreads();

#if HAVE_TDM
    if (wid == 0) {
      // ---- TDM: DMA K tile (64 rows x 64 bf16) into padded LDS rows ----
      const unsigned long long ga = (unsigned long long)(const void*)QKV
          + 2ull * ((baseRow + kb0) * 3072 + (size_t)h * 192 + 64);
      const unsigned int ldsa = (unsigned int)(unsigned long long)(const void*)&Ks[0];
      // D# group0: count=1 | lds_addr | global_addr[56:0] | type=2
      const v4u g0 = { 1u, ldsa, (unsigned int)ga,
                       (unsigned int)((ga >> 32) & 0x1FFFFFFull) | (2u << 30) };
      // D# group1: data_size=2B, pad_enable, pad_interval=32DW, pad_amount=2DW,
      //            tensor_dim0=3072, tensor_dim1=1024, tile 64x64, dim0_stride=3072
      const v8i g1 = { (int)((1u << 16) | (1u << 20) | (4u << 22) | (1u << 25)),
                       (int)(3072u << 16),
                       (int)(1024u << 16),
                       (int)(64u << 16),
                       64, 3072, 0, 0 };
      const v4i z4 = { 0, 0, 0, 0 };
      const v8i z8 = { 0, 0, 0, 0, 0, 0, 0, 0 };
      __builtin_amdgcn_tensor_load_to_lds(g0, g1, z4, z4, z8, 0);
    }
#else
    // ---- fallback: manual K tile copy ----
    #pragma unroll
    for (int i = 0; i < 4; ++i) {
      const int idx = tid + i * 256;
      const int sk  = idx >> 4;
      const int c4  = (idx & 15) * 4;
      *reinterpret_cast<uint2*>(&Ks[sk * 68 + c4]) =
          *reinterpret_cast<const uint2*>(
              QKV + (baseRow + kb0 + sk) * 3072 + h * 192 + 64 + c4);
    }
#endif

    // ---- stage V^T (2-byte transpose: not DMA-able), overlaps with TDM ----
    #pragma unroll
    for (int i = 0; i < 4; ++i) {
      const int idx = tid + i * 256;
      const int sk  = idx >> 4;
      const int c4  = (idx & 15) * 4;
      const uint2 vv = *reinterpret_cast<const uint2*>(
          QKV + (baseRow + kb0 + sk) * 3072 + h * 192 + 128 + c4);
      Vs[(c4 + 0) * 68 + sk] = (unsigned short)(vv.x & 0xFFFFu);
      Vs[(c4 + 1) * 68 + sk] = (unsigned short)(vv.x >> 16);
      Vs[(c4 + 2) * 68 + sk] = (unsigned short)(vv.y & 0xFFFFu);
      Vs[(c4 + 3) * 68 + sk] = (unsigned short)(vv.y >> 16);
    }

#if HAVE_TDM
    if (wid == 0) __builtin_amdgcn_s_wait_tensorcnt(0);
#endif
    __syncthreads();

    // ---- S = (Q K^T) * 0.125, diagonal -> -10 ----
    v8f st[4];
    #pragma unroll
    for (int t = 0; t < 4; ++t) {
      v8f s;
      #pragma unroll
      for (int r = 0; r < 8; ++r) s[r] = 0.0f;
      #pragma unroll
      for (int ks = 0; ks < 2; ++ks) {
        BF16Frag bk;
        #pragma unroll
        for (int v = 0; v < 8; ++v)
          bk.u[v] = *reinterpret_cast<const unsigned int*>(
              &Ks[(t * 16 + l16) * 68 + ks * 32 + ((v & 4) ? 16 : 0) + hf * 8 + 2 * (v & 3)]);
        s = __builtin_amdgcn_wmma_f32_16x16x32_bf16(
            false, qf[ks].v, false, bk.v, (short)0, s, false, false);
      }
      st[t] = s;
    }
    #pragma unroll
    for (int t = 0; t < 4; ++t)
      #pragma unroll
      for (int r = 0; r < 8; ++r) {
        const int qg = q0 + wid * 16 + hf * 8 + r;
        const int kg = kb0 + t * 16 + l16;
        st[t][r] = (qg == kg) ? -10.0f : st[t][r] * 0.125f;
      }

    // ---- online softmax (row reductions over 16-lane halves) ----
    #pragma unroll
    for (int r = 0; r < 8; ++r) {
      float vm = st[0][r];
      #pragma unroll
      for (int t = 1; t < 4; ++t) vm = fmaxf(vm, st[t][r]);
      vm = fmaxf(vm, __shfl_xor(vm, 1));
      vm = fmaxf(vm, __shfl_xor(vm, 2));
      vm = fmaxf(vm, __shfl_xor(vm, 4));
      vm = fmaxf(vm, __shfl_xor(vm, 8));
      const float newm = fmaxf(mrow[r], vm);
      const float corr = __expf(mrow[r] - newm);
      float rs = 0.0f;
      #pragma unroll
      for (int t = 0; t < 4; ++t) {
        const float p = __expf(st[t][r] - newm);
        st[t][r] = p;
        rs += p;
      }
      rs += __shfl_xor(rs, 1);
      rs += __shfl_xor(rs, 2);
      rs += __shfl_xor(rs, 4);
      rs += __shfl_xor(rs, 8);
      lrow[r] = lrow[r] * corr + rs;
      mrow[r] = newm;
      #pragma unroll
      for (int j = 0; j < 4; ++j) Oa[j][r] *= corr;
    }

    // ---- P to per-wave LDS (bf16), then O += P · V ----
    #pragma unroll
    for (int t = 0; t < 4; ++t)
      #pragma unroll
      for (int r = 0; r < 8; ++r)
        Pl[wid][(hf * 8 + r) * 68 + t * 16 + l16] = f2bf(st[t][r]);

    #pragma unroll
    for (int ks = 0; ks < 2; ++ks) {
      BF16Frag pa;
      #pragma unroll
      for (int v = 0; v < 8; ++v) {
        const int kk = ks * 32 + ((v & 4) ? 16 : 0) + hf * 8 + 2 * (v & 3);
        pa.u[v] = *reinterpret_cast<const unsigned int*>(&Pl[wid][l16 * 68 + kk]);
      }
      #pragma unroll
      for (int j = 0; j < 4; ++j) {
        BF16Frag bv;
        #pragma unroll
        for (int v = 0; v < 8; ++v)
          bv.u[v] = *reinterpret_cast<const unsigned int*>(
              &Vs[(j * 16 + l16) * 68 + ks * 32 + ((v & 4) ? 16 : 0) + hf * 8 + 2 * (v & 3)]);
        Oa[j] = __builtin_amdgcn_wmma_f32_16x16x32_bf16(
            false, pa.v, false, bv.v, (short)0, Oa[j], false, false);
      }
    }
  }

  // ---- finalize: Out = x + softmax(S)·V ----
  #pragma unroll
  for (int j = 0; j < 4; ++j) {
    const int gc = h * 64 + j * 16 + l16;
    #pragma unroll
    for (int r = 0; r < 8; ++r) {
      const int gs = q0 + wid * 16 + hf * 8 + r;
      const size_t o = (baseRow + gs) * 1024 + gc;
      Out[o] = X[o] + Oa[j][r] * (1.0f / lrow[r]);
    }
  }
}

// ---------------------------------------------------------------------------
extern "C" void kernel_launch(void* const* d_in, const int* in_sizes, int n_in,
                              void* d_out, int out_size, void* d_ws, size_t ws_size,
                              hipStream_t stream) {
  (void)in_sizes; (void)n_in; (void)out_size; (void)ws_size;
  const float* x      = (const float*)d_in[0];
  const float* w_qkv  = (const float*)d_in[1];
  const float* b_qkv  = (const float*)d_in[2];
  const float* w_fc   = (const float*)d_in[3];
  const float* b_fc   = (const float*)d_in[4];
  const float* w_proj = (const float*)d_in[5];
  const float* b_proj = (const float*)d_in[6];
  float* out = (float*)d_out;

  // workspace: QKV bf16 [8192,3072] then H bf16 [8192,4096]
  unsigned short* qkv_bf = (unsigned short*)d_ws;
  unsigned short* h_bf   = (unsigned short*)((char*)d_ws + (size_t)8192 * 3072 * 2);

  const dim3 blk(256);
  // 1) QKV projection -> bf16 (stays in L2)
  gemm_bf16_wmma<float, 0><<<dim3(24, 64), blk, 0, stream>>>(
      x, w_qkv, b_qkv, qkv_bf, nullptr, 8192, 3072, 1024);
  // 2) flash attention + residual -> d_out (fp32, doubles as MLP input)
  attn_flash_wmma<<<dim3(8, 16, 8), blk, 0, stream>>>(qkv_bf, x, out);
  // 3) FC + QuickGELU -> bf16
  gemm_bf16_wmma<float, 1><<<dim3(32, 64), blk, 0, stream>>>(
      out, w_fc, b_fc, h_bf, nullptr, 8192, 4096, 1024);
  // 4) Proj + bias + residual(d_out) -> d_out fp32
  gemm_bf16_wmma<unsigned short, 2><<<dim3(8, 64), blk, 0, stream>>>(
      h_bf, w_proj, b_proj, out, out, 8192, 1024, 4096);
}
